// LayoutLMv3CRF_38276748542519
// MI455X (gfx1250) — matrix operand order repaired and checked by reading, not compiled, not run on hardware
//
#include <hip/hip_runtime.h>
#include <math.h>

// ---------------------------------------------------------------------------
// LayoutLMv3 + CRF forward for MI455X (gfx1250, wave32, WMMA).
// GEMMs use v_wmma_f32_16x16x32_bf16 with on-the-fly fp32->bf16 conversion.
// Round 3: 128x128 block tile (8 waves x (64x32), 4x2 accumulators = 8 WMMAs
// per wave per K-step); uniform full-tile fast path with float4 global loads
// (no per-element exec-mask branching); LDS layouts keep both fragments as
// contiguous ds_load_b128 pairs.
// ---------------------------------------------------------------------------

#define B_     8
#define S_     512
#define H_     768
#define NLAY_  12
#define NH_    12
#define DH_    64
#define NL_    13
#define COORD_ 128
#define MAX2D_ 1024
#define FFH_   3072
#define TOK_   (B_ * S_)

typedef __attribute__((ext_vector_type(16))) __bf16 v16bf;
typedef __attribute__((ext_vector_type(8)))  float  v8f;

__device__ __forceinline__ unsigned short f2bf(float f) {
  unsigned u = __builtin_bit_cast(unsigned, f);
  u += 0x7FFFu + ((u >> 16) & 1u);              // round to nearest even
  return (unsigned short)(u >> 16);
}
__device__ __forceinline__ __bf16 bfbits(unsigned short u) {
  return __builtin_bit_cast(__bf16, u);
}
__device__ __forceinline__ float gelu_tanh(float x) {
  return 0.5f * x * (1.f + tanhf(0.7978845608028654f * (x + 0.044715f * x * x * x)));
}

// ---- block reductions (256 threads = 8 waves of 32) -----------------------
__device__ __forceinline__ float blockReduceSum256(float v) {
  __shared__ float red_s[8];
  __syncthreads();
#pragma unroll
  for (int o = 16; o > 0; o >>= 1) v += __shfl_down(v, o, 32);
  if ((threadIdx.x & 31) == 0) red_s[threadIdx.x >> 5] = v;
  __syncthreads();
  if (threadIdx.x < 32) {
    float t = (threadIdx.x < 8) ? red_s[threadIdx.x] : 0.f;
#pragma unroll
    for (int o = 4; o > 0; o >>= 1) t += __shfl_down(t, o, 32);
    if (threadIdx.x == 0) red_s[0] = t;
  }
  __syncthreads();
  return red_s[0];
}

__device__ __forceinline__ float blockReduceMax256(float v) {
  __shared__ float red_m[8];
  __syncthreads();
#pragma unroll
  for (int o = 16; o > 0; o >>= 1) v = fmaxf(v, __shfl_down(v, o, 32));
  if ((threadIdx.x & 31) == 0) red_m[threadIdx.x >> 5] = v;
  __syncthreads();
  if (threadIdx.x < 32) {
    float t = (threadIdx.x < 8) ? red_m[threadIdx.x] : -3.4e38f;
#pragma unroll
    for (int o = 4; o > 0; o >>= 1) t = fmaxf(t, __shfl_down(t, o, 32));
    if (threadIdx.x == 0) red_m[0] = t;
  }
  __syncthreads();
  return red_m[0];
}

// ---------------------------------------------------------------------------
// Batched WMMA GEMM: C = op(alpha * A@B + bias)
//   block = 256 threads = 8 waves (2 along M x 4 along N)
//   block tile 128(M) x 128(N), K stepped by 32
//   each wave: 64x32 = 4x2 of 16x16x32 bf16 WMMA, f32 accumulate
// LDS layouts (row pitch 40 shorts = 80 B, 16 B aligned):
//   As[m][k]  -> A fragment = two contiguous 16 B chunks per lane
//   Bs[n][k]  -> B fragment = one contiguous 32 B run per lane
// ---------------------------------------------------------------------------
#define BMT 128
#define BNT 128
#define BKT 32
#define KPAD 8

__global__ __launch_bounds__(256) void wmma_gemm_kernel(
    const float* __restrict__ A, const float* __restrict__ Bm,
    const float* __restrict__ bias, float* __restrict__ C,
    int M, int N, int K, int lda, int ldb, int ldc,
    int transB, float alpha, int op, int innerCnt,
    long long sAo, long long sAi, long long sBo, long long sBi,
    long long sCo, long long sCi)
{
  __shared__ unsigned short As[BMT][BKT + KPAD];   // 10 KB
  __shared__ unsigned short Bs[BNT][BKT + KPAD];   // 10 KB

  int g = blockIdx.z;
  int go = g / innerCnt, gi = g - go * innerCnt;
  A  += (long long)go * sAo + (long long)gi * sAi;
  Bm += (long long)go * sBo + (long long)gi * sBi;
  C  += (long long)go * sCo + (long long)gi * sCi;

  const int tid = threadIdx.x;
  const int wave = tid >> 5, lane = tid & 31;
  const int wm = wave & 1, wn = wave >> 1;     // 2 x 4 wave grid
  const int laneRow = lane & 15, laneHi = lane >> 4;
  const int rowBase = blockIdx.y * BMT;
  const int colBase = blockIdx.x * BNT;

  const bool full = (rowBase + BMT <= M) && (colBase + BNT <= N) &&
                    ((K & (BKT - 1)) == 0);

  v8f acc[4][2] = {};

  for (int k0 = 0; k0 < K; k0 += BKT) {
    if (full) {
      // ---- fast path: straight-line float4 loads -------------------------
#pragma unroll
      for (int u = 0; u < 4; ++u) {            // A tile 128x32 = 1024 float4
        int i = tid + u * 256;
        int r = i >> 3, c4 = i & 7;
        const float4 f = *reinterpret_cast<const float4*>(
            &A[(long long)(rowBase + r) * lda + k0 + c4 * 4]);
        As[r][c4 * 4 + 0] = f2bf(f.x);
        As[r][c4 * 4 + 1] = f2bf(f.y);
        As[r][c4 * 4 + 2] = f2bf(f.z);
        As[r][c4 * 4 + 3] = f2bf(f.w);
      }
      if (!transB) {
#pragma unroll
        for (int u = 0; u < 4; ++u) {          // B tile 32x128, n fastest
          int i = tid + u * 256;
          int r = i >> 5, c4 = i & 31;
          const float4 f = *reinterpret_cast<const float4*>(
              &Bm[(long long)(k0 + r) * ldb + colBase + c4 * 4]);
          Bs[c4 * 4 + 0][r] = f2bf(f.x);
          Bs[c4 * 4 + 1][r] = f2bf(f.y);
          Bs[c4 * 4 + 2][r] = f2bf(f.z);
          Bs[c4 * 4 + 3][r] = f2bf(f.w);
        }
      } else {
#pragma unroll
        for (int u = 0; u < 4; ++u) {          // B tile 128xK, k fastest
          int i = tid + u * 256;
          int c = i >> 3, k4 = i & 7;
          const float4 f = *reinterpret_cast<const float4*>(
              &Bm[(long long)(colBase + c) * ldb + k0 + k4 * 4]);
          Bs[c][k4 * 4 + 0] = f2bf(f.x);
          Bs[c][k4 * 4 + 1] = f2bf(f.y);
          Bs[c][k4 * 4 + 2] = f2bf(f.z);
          Bs[c][k4 * 4 + 3] = f2bf(f.w);
        }
      }
      if (k0 + BKT < K) {                      // uniform-guard prefetch
        __builtin_prefetch(&A[(long long)(rowBase + (tid >> 3)) * lda + k0 + BKT], 0, 0);
        if (!transB)
          __builtin_prefetch(&Bm[(long long)(k0 + BKT + (tid >> 5)) * ldb + colBase + (tid & 31) * 4], 0, 0);
        else
          __builtin_prefetch(&Bm[(long long)(colBase + (tid >> 3)) * ldb + k0 + BKT], 0, 0);
      }
    } else {
      // ---- slow path: per-element bounds-checked loads -------------------
      for (int i = tid; i < BMT * BKT; i += 256) {
        int r = i >> 5, c = i & 31;
        int gr = rowBase + r, gc = k0 + c;
        float f = (gr < M && gc < K) ? A[(long long)gr * lda + gc] : 0.f;
        As[r][c] = f2bf(f);
      }
      if (!transB) {
        for (int i = tid; i < BKT * BNT; i += 256) {
          int r = i >> 7, c = i & 127;         // r = k, c = n
          int gk = k0 + r, gn = colBase + c;
          float f = (gk < K && gn < N) ? Bm[(long long)gk * ldb + gn] : 0.f;
          Bs[c][r] = f2bf(f);
        }
      } else {
        for (int i = tid; i < BKT * BNT; i += 256) {
          int c = i >> 5, r = i & 31;          // c = n, r = k
          int gk = k0 + r, gn = colBase + c;
          float f = (gk < K && gn < N) ? Bm[(long long)gn * ldb + gk] : 0.f;
          Bs[c][r] = f2bf(f);
        }
      }
    }
    __syncthreads();

    // assemble fragments in k = 16*laneHi + e order (same for A and B) and issue 4x2 WMMAs
    v16bf afr[4], bfr[2];
#pragma unroll
    for (int mi = 0; mi < 4; ++mi) {
      int ar = wm * 64 + mi * 16 + laneRow;
#pragma unroll
      for (int e = 0; e < 16; ++e) {
        int ka = laneHi * 16 + e;                      // one 32 B run
        afr[mi][e] = bfbits(As[ar][ka]);
      }
    }
#pragma unroll
    for (int ni = 0; ni < 2; ++ni) {
      int bcol = wn * 32 + ni * 16 + laneRow;
#pragma unroll
      for (int e = 0; e < 16; ++e) {
        int kb = laneHi * 16 + e;                      // one 32 B run
        bfr[ni][e] = bfbits(Bs[bcol][kb]);
      }
    }
#pragma unroll
    for (int mi = 0; mi < 4; ++mi)
#pragma unroll
      for (int ni = 0; ni < 2; ++ni)
        acc[mi][ni] = __builtin_amdgcn_wmma_f32_16x16x32_bf16(
            false, afr[mi], false, bfr[ni], (short)0, acc[mi][ni], false, false);
    __syncthreads();
  }

  // epilogue: C/D layout M = r + 8*laneHi, N = lane&15
#pragma unroll
  for (int mi = 0; mi < 4; ++mi) {
#pragma unroll
    for (int ni = 0; ni < 2; ++ni) {
      int col = colBase + wn * 32 + ni * 16 + laneRow;
      if (col < N) {
        float bv = bias ? bias[col] : 0.f;
#pragma unroll
        for (int r = 0; r < 8; ++r) {
          int row = rowBase + wm * 64 + mi * 16 + r + 8 * laneHi;
          if (row < M) {
            float v = acc[mi][ni][r] * alpha + bv;
            if (op == 1) v = gelu_tanh(v);
            C[(long long)row * ldc + col] = v;
          }
        }
      }
    }
  }
}

// ---------------------------------------------------------------------------
// Embedding (word + 1D pos + 6x concat 2D spatial) + LayerNorm. 1 block/token.
// ---------------------------------------------------------------------------
__global__ __launch_bounds__(256) void embed_ln_kernel(
    const int* __restrict__ ids, const int* __restrict__ bbox,
    const float* __restrict__ wemb, const float* __restrict__ pemb,
    const float* __restrict__ xe, const float* __restrict__ ye,
    const float* __restrict__ he, const float* __restrict__ we,
    const float* __restrict__ gam, const float* __restrict__ bet,
    float* __restrict__ out)
{
  int t = blockIdx.x;
  int s = t % S_;
  int id = ids[t];
  int x1 = bbox[t * 4 + 0], y1 = bbox[t * 4 + 1];
  int x2 = bbox[t * 4 + 2], y2 = bbox[t * 4 + 3];
  int hh = y2 - y1; hh = hh < 0 ? 0 : (hh > MAX2D_ - 1 ? MAX2D_ - 1 : hh);
  int ww = x2 - x1; ww = ww < 0 ? 0 : (ww > MAX2D_ - 1 ? MAX2D_ - 1 : ww);
  const float* segp[6] = {
      xe + (long long)x1 * COORD_, ye + (long long)y1 * COORD_,
      xe + (long long)x2 * COORD_, ye + (long long)y2 * COORD_,
      he + (long long)hh * COORD_, we + (long long)ww * COORD_};

  float v[3];
#pragma unroll
  for (int i = 0; i < 3; ++i) {
    int h = threadIdx.x + i * 256;
    v[i] = wemb[(long long)id * H_ + h] + pemb[s * H_ + h] + segp[h >> 7][h & 127];
  }
  float mu = blockReduceSum256(v[0] + v[1] + v[2]) * (1.f / H_);
  float sq = 0.f;
#pragma unroll
  for (int i = 0; i < 3; ++i) { float d = v[i] - mu; sq += d * d; }
  float inv = rsqrtf(blockReduceSum256(sq) * (1.f / H_) + 1e-5f);
#pragma unroll
  for (int i = 0; i < 3; ++i) {
    int h = threadIdx.x + i * 256;
    out[(long long)t * H_ + h] = (v[i] - mu) * inv * gam[h] + bet[h];
  }
}

// ---------------------------------------------------------------------------
// out = LN(xin + delta) * gamma + beta   (in-place safe: out may alias xin)
// ---------------------------------------------------------------------------
__global__ __launch_bounds__(256) void resid_ln_kernel(
    const float* __restrict__ xin, const float* __restrict__ delta,
    const float* __restrict__ gam, const float* __restrict__ bet,
    float* __restrict__ out)
{
  long long t = blockIdx.x;
  float v[3];
#pragma unroll
  for (int i = 0; i < 3; ++i) {
    int h = threadIdx.x + i * 256;
    v[i] = xin[t * H_ + h] + delta[t * H_ + h];
  }
  float mu = blockReduceSum256(v[0] + v[1] + v[2]) * (1.f / H_);
  float sq = 0.f;
#pragma unroll
  for (int i = 0; i < 3; ++i) { float d = v[i] - mu; sq += d * d; }
  float inv = rsqrtf(blockReduceSum256(sq) * (1.f / H_) + 1e-5f);
#pragma unroll
  for (int i = 0; i < 3; ++i) {
    int h = threadIdx.x + i * 256;
    out[t * H_ + h] = (v[i] - mu) * inv * gam[h] + bet[h];
  }
}

// ---------------------------------------------------------------------------
// masked softmax over the last dim of scores[B*NH, S, S]; 1 block per row
// ---------------------------------------------------------------------------
__global__ __launch_bounds__(256) void softmax_bias_kernel(
    float* __restrict__ sc, const int* __restrict__ mask)
{
  long long row = blockIdx.x;                 // g*S + q
  int gidx = blockIdx.x / S_;
  int b = gidx / NH_;
  float* p = sc + row * S_;
  float v[2];
#pragma unroll
  for (int i = 0; i < 2; ++i) {
    int k = threadIdx.x + i * 256;
    v[i] = p[k] + (mask[b * S_ + k] ? 0.f : -1e9f);
  }
  float mx = blockReduceMax256(fmaxf(v[0], v[1]));
  float e0 = expf(v[0] - mx), e1 = expf(v[1] - mx);
  float inv = 1.f / blockReduceSum256(e0 + e1);
  p[threadIdx.x] = e0 * inv;
  p[threadIdx.x + 256] = e1 * inv;
}

// ---------------------------------------------------------------------------
// CRF: gold score + forward recursion (logZ). 1 block (32 threads) per batch.
// lossB[b] = logZ_b - score_b
// ---------------------------------------------------------------------------
__global__ __launch_bounds__(32) void crf_kernel(
    const float* __restrict__ em,   // [B*S, NL]
    const int* __restrict__ mask, const int* __restrict__ labels,
    const float* __restrict__ trans, const float* __restrict__ startv,
    const float* __restrict__ endv, float* __restrict__ lossB)
{
  int b = blockIdx.x;
  int j = threadIdx.x;
  __shared__ float alpha[NL_];
  __shared__ float sc_sh;

  if (j < NL_) alpha[j] = startv[j] + em[(long long)b * S_ * NL_ + j];
  if (j == 0) {
    int l0 = labels[b * S_];
    float sc = startv[l0] + em[(long long)b * S_ * NL_ + l0];
    int prev = l0;
    for (int t = 1; t < S_; ++t) {
      int lab = labels[b * S_ + t];
      if (mask[b * S_ + t])
        sc += em[((long long)b * S_ + t) * NL_ + lab] + trans[prev * NL_ + lab];
      prev = lab;
    }
    int last = -1;
    for (int t = 0; t < S_; ++t) last += mask[b * S_ + t];
    sc += endv[labels[b * S_ + last]];
    sc_sh = sc;
  }
  __syncthreads();

  for (int t = 1; t < S_; ++t) {
    float nv = 0.f;
    if (j < NL_) {
      float mx = -3.4e38f;
      for (int i = 0; i < NL_; ++i)
        mx = fmaxf(mx, alpha[i] + trans[i * NL_ + j]);
      float s = 0.f;
      for (int i = 0; i < NL_; ++i)
        s += expf(alpha[i] + trans[i * NL_ + j] - mx);
      nv = mx + logf(s) + em[((long long)b * S_ + t) * NL_ + j];
    }
    __syncthreads();
    if (j < NL_ && mask[b * S_ + t]) alpha[j] = nv;
    __syncthreads();
  }

  if (j == 0) {
    float mx = -3.4e38f;
    for (int i = 0; i < NL_; ++i) mx = fmaxf(mx, alpha[i] + endv[i]);
    float s = 0.f;
    for (int i = 0; i < NL_; ++i) s += expf(alpha[i] + endv[i] - mx);
    lossB[b] = (mx + logf(s)) - sc_sh;
  }
}

__global__ void loss_mean_kernel(const float* __restrict__ lossB,
                                 float* __restrict__ out)
{
  if (threadIdx.x == 0 && blockIdx.x == 0) {
    float s = 0.f;
    for (int b = 0; b < B_; ++b) s += lossB[b];
    out[0] = s * (1.f / B_);
  }
}

// ---------------------------------------------------------------------------
static inline void launch_gemm(hipStream_t stream,
    const float* A, const float* Bm, const float* bias, float* C,
    int M, int N, int K, int lda, int ldb, int ldc,
    int transB, float alpha, int op,
    int batch = 1, int innerCnt = 1,
    long long sAo = 0, long long sAi = 0,
    long long sBo = 0, long long sBi = 0,
    long long sCo = 0, long long sCi = 0)
{
  dim3 grid((N + BNT - 1) / BNT, (M + BMT - 1) / BMT, batch);
  wmma_gemm_kernel<<<grid, 256, 0, stream>>>(A, Bm, bias, C, M, N, K,
      lda, ldb, ldc, transB, alpha, op, innerCnt, sAo, sAi, sBo, sBi, sCo, sCi);
}

// Assumed d_in layout: setup_inputs() insertion order, params flattened in
// source order (word_emb, pos_emb, x/y/h/w_emb, ln_g, ln_b, layers{Wq,bq,Wk,
// bk,Wv,bv,Wo,bo,g1,b1,W1,bf1,W2,bf2,g2,b2}, Wc, bc, trans, start, end).
extern "C" void kernel_launch(void* const* d_in, const int* in_sizes, int n_in,
                              void* d_out, int out_size, void* d_ws, size_t ws_size,
                              hipStream_t stream)
{
  (void)in_sizes; (void)n_in; (void)out_size; (void)ws_size;

  const int* ids    = (const int*)d_in[0];
  const int* bbox   = (const int*)d_in[1];
  const int* amask  = (const int*)d_in[2];
  const int* labels = (const int*)d_in[3];
  const float* PWORD = (const float*)d_in[4];
  const float* PPOS  = (const float*)d_in[5];
  const float* PXE   = (const float*)d_in[6];
  const float* PYE   = (const float*)d_in[7];
  const float* PHE   = (const float*)d_in[8];
  const float* PWE   = (const float*)d_in[9];
  const float* PLNG  = (const float*)d_in[10];
  const float* PLNB  = (const float*)d_in[11];
  const float* PWQ  = (const float*)d_in[12];
  const float* PBQ  = (const float*)d_in[13];
  const float* PWK  = (const float*)d_in[14];
  const float* PBK  = (const float*)d_in[15];
  const float* PWV  = (const float*)d_in[16];
  const float* PBV  = (const float*)d_in[17];
  const float* PWO  = (const float*)d_in[18];
  const float* PBO  = (const float*)d_in[19];
  const float* PG1  = (const float*)d_in[20];
  const float* PB1  = (const float*)d_in[21];
  const float* PW1  = (const float*)d_in[22];
  const float* PBF1 = (const float*)d_in[23];
  const float* PW2  = (const float*)d_in[24];
  const float* PBF2 = (const float*)d_in[25];
  const float* PG2  = (const float*)d_in[26];
  const float* PB2  = (const float*)d_in[27];
  const float* PWC  = (const float*)d_in[28];
  const float* PBC  = (const float*)d_in[29];
  const float* PTR  = (const float*)d_in[30];
  const float* PST  = (const float*)d_in[31];
  const float* PEN  = (const float*)d_in[32];

  // workspace carve (~227 MB of f32)
  float* ws     = (float*)d_ws;
  float* x      = ws;                                  // [TOK,H]
  float* q      = x    + (size_t)TOK_ * H_;            // [TOK,H]
  float* kbuf   = q    + (size_t)TOK_ * H_;            // [TOK,H]
  float* vbuf   = kbuf + (size_t)TOK_ * H_;            // [TOK,H]
  float* tmp    = vbuf + (size_t)TOK_ * H_;            // [TOK,H]
  float* ctx    = tmp  + (size_t)TOK_ * H_;            // [TOK,H]
  float* ffn    = ctx  + (size_t)TOK_ * H_;            // [TOK,FFH]
  float* scores = ffn  + (size_t)TOK_ * FFH_;          // [B*NH,S,S]
  float* em     = scores + (size_t)B_ * NH_ * S_ * S_; // [TOK,NL]
  float* lossB  = em   + (size_t)TOK_ * NL_;           // [B]

  embed_ln_kernel<<<TOK_, 256, 0, stream>>>(
      ids, bbox, PWORD, PPOS, PXE, PYE, PHE, PWE, PLNG, PLNB, x);

  const long long sBH = (long long)S_ * H_;   // batch(b) stride in [B,S,H]
  const long long sSS = (long long)S_ * S_;   // per-head stride in scores

  for (int l = 0; l < NLAY_; ++l) {
    const float* Wq = PWQ + (size_t)l * H_ * H_;
    const float* Wk = PWK + (size_t)l * H_ * H_;
    const float* Wv = PWV + (size_t)l * H_ * H_;
    const float* Wo = PWO + (size_t)l * H_ * H_;
    const float* W1 = PW1 + (size_t)l * H_ * FFH_;
    const float* W2 = PW2 + (size_t)l * FFH_ * H_;
    const float* bq = PBQ + (size_t)l * H_;
    const float* bk = PBK + (size_t)l * H_;
    const float* bv = PBV + (size_t)l * H_;
    const float* bo = PBO + (size_t)l * H_;
    const float* bf1 = PBF1 + (size_t)l * FFH_;
    const float* bf2 = PBF2 + (size_t)l * H_;
    const float* g1 = PG1 + (size_t)l * H_;
    const float* b1 = PB1 + (size_t)l * H_;
    const float* g2 = PG2 + (size_t)l * H_;
    const float* b2 = PB2 + (size_t)l * H_;

    // QKV projections
    launch_gemm(stream, x, Wq, bq, q,    TOK_, H_, H_, H_, H_, H_, 0, 1.f, 0);
    launch_gemm(stream, x, Wk, bk, kbuf, TOK_, H_, H_, H_, H_, H_, 0, 1.f, 0);
    launch_gemm(stream, x, Wv, bv, vbuf, TOK_, H_, H_, H_, H_, H_, 0, 1.f, 0);

    // scores[b,h] = (Q_bh @ K_bh^T) / sqrt(DH)   (batched over B*NH)
    launch_gemm(stream, q, kbuf, nullptr, scores,
                S_, S_, DH_, H_, H_, S_, /*transB=*/1, 0.125f, 0,
                B_ * NH_, NH_, sBH, DH_, sBH, DH_,
                (long long)NH_ * sSS, sSS);

    softmax_bias_kernel<<<B_ * NH_ * S_, 256, 0, stream>>>(scores, amask);

    // ctx[b,h] = P_bh @ V_bh -> lands directly in [B,S,H] layout
    launch_gemm(stream, scores, vbuf, nullptr, ctx,
                S_, DH_, S_, S_, H_, H_, 0, 1.f, 0,
                B_ * NH_, NH_, (long long)NH_ * sSS, sSS, sBH, DH_,
                sBH, DH_);

    launch_gemm(stream, ctx, Wo, bo, tmp, TOK_, H_, H_, H_, H_, H_, 0, 1.f, 0);
    resid_ln_kernel<<<TOK_, 256, 0, stream>>>(x, tmp, g1, b1, x);

    launch_gemm(stream, x,   W1, bf1, ffn, TOK_, FFH_, H_,  H_,  FFH_, FFH_, 0, 1.f, 1);
    launch_gemm(stream, ffn, W2, bf2, tmp, TOK_, H_,  FFH_, FFH_, H_,  H_,  0, 1.f, 0);
    resid_ln_kernel<<<TOK_, 256, 0, stream>>>(x, tmp, g2, b2, x);
  }

  // classifier -> emissions [TOK, NL]
  launch_gemm(stream, x, PWC, PBC, em, TOK_, NL_, H_, H_, NL_, NL_, 0, 1.f, 0);

  crf_kernel<<<B_, 32, 0, stream>>>(em, amask, labels, PTR, PST, PEN, lossB);
  loss_mean_kernel<<<1, 32, 0, stream>>>(lossB, (float*)d_out);
}
